// ConvUnit_39135742001480
// MI455X (gfx1250) — compile-verified, matrix operand order checked by reading, not run on hardware
//
#include <hip/hip_runtime.h>

typedef __attribute__((ext_vector_type(16))) _Float16 v16h;
typedef __attribute__((ext_vector_type(8)))  float    v8f;

#define CIN   64
#define COUT  128
#define HIN   112
#define WIN   112
#define HOUT  110
#define WOUT  110
#define NBATCH 16
#define NCHUNK 18          // K = 576 = 18 chunks of 32
#define QTILES 7           // ceil(110/16)
#define WFRAG_HALFS (8 * NCHUNK * 32 * 16)   // [mtile][chunk][lane][16 halfs]

// ---------------------------------------------------------------------------
// Prep: swizzle fp32 OIHW weights into wave32 A-fragment layout, split hi/lo.
// K ordering: k = (r*3+s)*64 + c  ->  chunk t: rs = t>>1, c = 32*(t&1) + klo
// A-fragment (16-bit A 16x32): lane<16 holds K {0..7,16..23}; lane>=16 holds
// K {8..15,24..31}; within a lane, half m maps klo = (m<8 ? m : m+8).
// ---------------------------------------------------------------------------
__global__ void prep_weights_kernel(const float* __restrict__ w,
                                    _Float16* __restrict__ whi,
                                    _Float16* __restrict__ wlo) {
    int idx = blockIdx.x * blockDim.x + threadIdx.x;
    if (idx >= WFRAG_HALFS) return;
    int m    = idx & 15;
    int lane = (idx >> 4) & 31;
    int t    = (idx >> 9) % NCHUNK;
    int mt   = idx / (NCHUNK * 32 * 16);

    int row  = lane & 15;
    int cout = mt * 16 + row;
    int klo  = ((lane >> 4) << 3) + ((m < 8) ? m : m + 8);
    int rs   = t >> 1;
    int c    = ((t & 1) << 5) + klo;

    float wv = w[(cout * CIN + c) * 9 + rs];     // OIHW, rs = r*3+s
    _Float16 hi = (_Float16)wv;
    whi[idx] = hi;
    wlo[idx] = (_Float16)(wv - (float)hi);
}

// ---------------------------------------------------------------------------
// Main: fused 8-bit-plane implicit-GEMM conv with per-plane quantization.
// One wave computes one 16(cout) x 16(q) tile for a given (n, p, qtile).
// Block = 4 waves = 64 couts; grid.y picks the cout half.
// B-fragments (bit-plane activations) are identical across the 4 waves, so
// they are built cooperatively once per block in double-buffered LDS:
//   lds[buf][plane][pair j][lane] : uint2 = {u[2j], u[2j+1]} of the v16h frag
// Wave w owns halves m = 4w..4w+3 (= u[2w], u[2w+1] = uint2 j .. = w) for all
// 8 planes: 4 global loads + 8 ds_store_b64 per wave per chunk.
// ---------------------------------------------------------------------------
__global__ __launch_bounds__(128)
void conv_bitplane_wmma_kernel(const float* __restrict__ x,
                               const _Float16* __restrict__ whi,
                               const _Float16* __restrict__ wlo,
                               const float* __restrict__ bias,
                               float* __restrict__ out) {
    __shared__ unsigned ldsb[2][8][4][32][2];   // 16 KB, bank-conflict-free

    const int lane = threadIdx.x & 31;
    const int wave = threadIdx.x >> 5;

    const int bx = blockIdx.x;
    const int qt = bx % QTILES;
    const int p  = (bx / QTILES) % HOUT;
    const int n  = bx / (QTILES * HOUT);
    const int mtile = blockIdx.y * 4 + wave;

    const int j     = lane & 15;   // output column within tile (B N-index)
    const int hk    = lane >> 4;   // lane half
    const int qbase = qt * 16;

    const float* __restrict__ xslab = x + (size_t)n * CIN * HIN * WIN;

    v8f zero = {};
    v8f acc[8];
#pragma unroll
    for (int i = 0; i < 8; ++i) acc[i] = zero;

    const v16h* __restrict__ afrag_hi =
        (const v16h*)(whi + (size_t)mtile * NCHUNK * 32 * 16);
    const v16h* __restrict__ afrag_lo =
        (const v16h*)(wlo + (size_t)mtile * NCHUNK * 32 * 16);

    // ---- cooperative B-fragment builder (wave 'wave' owns uint2 index j=wave)
    auto build_chunk = [&](int t, int buf) {
        const int rs = t >> 1;
        const int r  = rs / 3;
        const int s  = rs - r * 3;
        const int cbase = ((t & 1) << 5) + (hk << 3);

        int col = qbase + j + s;            // input column; clamp keeps edge
        if (col > WIN - 1) col = WIN - 1;   // in-bounds (masked at store)
        const float* __restrict__ xb = xslab + (size_t)(p + r) * WIN + col;

        int av[4];
#pragma unroll
        for (int q = 0; q < 4; ++q) {
            int m   = wave * 4 + q;               // fragment half index
            int klo = (m < 8) ? m : m + 8;
            int c   = cbase + klo;
            float xf = xb[(size_t)c * (HIN * WIN)];
            float cl = fminf(fmaxf(xf, -128.0f), 127.0f);
            av[q] = ((int)cl) & 0xFF;             // trunc toward zero -> byte
        }
#pragma unroll
        for (int i = 0; i < 8; ++i) {
            unsigned u0 = ((unsigned)(av[0] >> i) & 1u) * 0x3C00u |
                          ((unsigned)(av[1] >> i) & 1u) * 0x3C000000u;
            unsigned u1 = ((unsigned)(av[2] >> i) & 1u) * 0x3C00u |
                          ((unsigned)(av[3] >> i) & 1u) * 0x3C000000u;
            *(uint2*)&ldsb[buf][i][wave][lane][0] = make_uint2(u0, u1);
        }
    };

    build_chunk(0, 0);

    for (int t = 0; t < NCHUNK; ++t) {
        __syncthreads();                      // builds of chunk t visible
        if (t + 1 < NCHUNK) build_chunk(t + 1, (t + 1) & 1);

        v16h ahi = afrag_hi[t * 32 + lane];
        v16h alo = afrag_lo[t * 32 + lane];
        const int buf = t & 1;

#pragma unroll
        for (int i = 0; i < 8; ++i) {
            union { uint2 v[4]; v16h h; } b;
#pragma unroll
            for (int jj = 0; jj < 4; ++jj)
                b.v[jj] = *(const uint2*)&ldsb[buf][i][jj][lane][0];
            acc[i] = __builtin_amdgcn_wmma_f32_16x16x32_f16(
                false, ahi, false, b.h, (short)0, acc[i], false, false);
            acc[i] = __builtin_amdgcn_wmma_f32_16x16x32_f16(
                false, alo, false, b.h, (short)0, acc[i], false, false);
        }
    }

    // Epilogue: per-plane quantization, weighted recombination, bias, store.
    const int qcol = qbase + j;
    if (qcol < WOUT) {
#pragma unroll
        for (int rrow = 0; rrow < 8; ++rrow) {
            float sum = 0.0f;
#pragma unroll
            for (int i = 0; i < 8; ++i) {
                float q = rintf(acc[i][rrow] * 0.125f);        // round(y/8), RTE
                q = fminf(fmaxf(q, -128.0f), 127.0f);
                float wi = (i == 7) ? -1024.0f : (float)(8 << i); // 8*(+-2^i)
                sum += q * wi;
            }
            int cout = mtile * 16 + rrow + (hk << 3);   // C-layout row mapping
            size_t oidx = (((size_t)n * COUT + cout) * HOUT + p) * WOUT + qcol;
            out[oidx] = sum + bias[cout];
        }
    }
}

extern "C" void kernel_launch(void* const* d_in, const int* in_sizes, int n_in,
                              void* d_out, int out_size, void* d_ws, size_t ws_size,
                              hipStream_t stream) {
    const float* x      = (const float*)d_in[0];   // (16,64,112,112)
    const float* weight = (const float*)d_in[1];   // (128,64,3,3)
    const float* bias   = (const float*)d_in[2];   // (128,)
    float* out = (float*)d_out;                    // (16,128,110,110)

    // Workspace: hi halfs then lo halfs (2 * 73728 * 2 B = 294912 B).
    _Float16* whi = (_Float16*)d_ws;
    _Float16* wlo = whi + WFRAG_HALFS;

    {
        int total = WFRAG_HALFS;
        int blk = 256;
        prep_weights_kernel<<<(total + blk - 1) / blk, blk, 0, stream>>>(
            weight, whi, wlo);
    }
    {
        dim3 grid(QTILES * HOUT * NBATCH, 2, 1);   // 12320 x 2 blocks
        dim3 block(128, 1, 1);                     // 4 waves
        conv_bitplane_wmma_kernel<<<grid, block, 0, stream>>>(
            x, whi, wlo, bias, out);
    }
}
